// MultiHeadAttention_13855564497010
// MI455X (gfx1250) — matrix operand order repaired and checked by reading, not compiled, not run on hardware
//
#include <hip/hip_runtime.h>
#include <hip/hip_bf16.h>

typedef _Float16 v16h __attribute__((ext_vector_type(16)));
typedef _Float16 v8h  __attribute__((ext_vector_type(8)));
typedef _Float16 v4h  __attribute__((ext_vector_type(4)));
typedef float    v8f  __attribute__((ext_vector_type(8)));
typedef float    v4f  __attribute__((ext_vector_type(4)));

#define B_  4
#define S_  2048
#define H_  16
#define DK  64
#define D_  1024
#define M_  8192   // B_*S_

union F16x16 { v16h v; v8h h[2]; };

__device__ __forceinline__ v8f wmma_f16(v16h a, v16h b, v8f c) {
  // D = A(16x32 f16) x B(32x16 f16) + C(16x16 f32)
  return __builtin_amdgcn_wmma_f32_16x16x32_f16(
      /*neg_a=*/false, a, /*neg_b=*/false, b,
      /*c_mod=*/(short)0, c, /*reuse_a=*/false, /*reuse_b=*/false);
}

// Async copy 16B global -> LDS (ASYNCcnt-tracked; CDNA5 GLOBAL_LOAD_ASYNC_TO_LDS_B128)
__device__ __forceinline__ void async_ld_b128(const _Float16* g, const _Float16* l) {
  unsigned loff = (unsigned)(size_t)l;  // LDS flat addr low 32 bits = LDS offset
  asm volatile("global_load_async_to_lds_b128 %0, %1, off"
               :: "v"(loff), "v"(g) : "memory");
}
__device__ __forceinline__ void wait_async_le2() {
  asm volatile("s_wait_asynccnt 0x2" ::: "memory");
}
__device__ __forceinline__ void wait_async_0() {
  asm volatile("s_wait_asynccnt 0x0" ::: "memory");
}

__device__ __forceinline__ float red_max16(float v) {
  // xor masks < 16 stay inside each 16-lane half (wave32)
  v = fmaxf(v, __shfl_xor(v, 1, 32));
  v = fmaxf(v, __shfl_xor(v, 2, 32));
  v = fmaxf(v, __shfl_xor(v, 4, 32));
  v = fmaxf(v, __shfl_xor(v, 8, 32));
  return v;
}
__device__ __forceinline__ float red_sum16(float v) {
  v += __shfl_xor(v, 1, 32);
  v += __shfl_xor(v, 2, 32);
  v += __shfl_xor(v, 4, 32);
  v += __shfl_xor(v, 8, 32);
  return v;
}

// ---------------- f32 -> f16 conversion (vectorized x4) ----------------
__global__ void cvt_kernel(const float* __restrict__ src,
                           _Float16* __restrict__ dst, int n4) {
  int i = blockIdx.x * blockDim.x + threadIdx.x;
  int stride = gridDim.x * blockDim.x;
  const v4f* s4 = (const v4f*)src;
  v4h* d4 = (v4h*)dst;
  for (; i < n4; i += stride) {
    v4f x = s4[i];
    v4h y;
    y.x = (_Float16)x.x; y.y = (_Float16)x.y;
    y.z = (_Float16)x.z; y.w = (_Float16)x.w;
    d4[i] = y;
  }
}

// ---------------- Y = X @ W.T  (X: [M,1024] f16, W: [1024,1024] f16) ----
// 8 waves / block, block tile 128(M) x 128(N); wave tile 64 x 32 (4x2 WMMA tiles).
// W tile (128 rows x 32 k = 8KB) is shared by all waves -> staged in LDS via
// async loads, double-buffered. A frags direct from global (WGP$ absorbs 4x reuse).
// mode 0: f16 -> head-split  [B,H,S,dk]   (Q, K)
// mode 1: f16 -> head-transp [B,H,dk,S]   (V, so PV B-frags are contiguous)
// mode 2: f32 -> row-major   [M,1024]     (final output)
#define BSTRIDE 40   // padded row stride (halves): conflict-free ds reads
__global__ __launch_bounds__(256) void gemm_xwt(const _Float16* __restrict__ X,
                                                const _Float16* __restrict__ W,
                                                void* __restrict__ Y, int mode) {
  __shared__ _Float16 sB[2][128 * BSTRIDE];
  int tid  = threadIdx.x;
  int lane = tid & 31;
  int w    = tid >> 5;
  int wm = w >> 2, wn = w & 3;
  int m0   = blockIdx.y * 128 + wm * 64;
  int nblk = blockIdx.x * 128;
  int ml = lane & 15, hi = lane >> 4;

  // cooperative async stage of W rows [nblk, nblk+128) x k-chunk [kb, kb+32)
  auto stageB = [&](int buf, int kb) {
#pragma unroll
    for (int r = 0; r < 2; r++) {
      int c = tid + 256 * r;       // 512 chunks of 8 halves (16B)
      int row = c >> 2, ch = c & 3;
      async_ld_b128(W + (size_t)(nblk + row) * D_ + kb + ch * 8,
                    &sB[buf][row * BSTRIDE + ch * 8]);
    }
  };

  v8f acc[4][2];
#pragma unroll
  for (int mi = 0; mi < 4; mi++)
#pragma unroll
    for (int ni = 0; ni < 2; ni++) acc[mi][ni] = {};

  stageB(0, 0);
  int buf = 0;
  for (int kb = 0; kb < D_; kb += 32, buf ^= 1) {
    bool pre = (kb + 32 < D_);
    if (pre) stageB(buf ^ 1, kb + 32);
    if (pre) wait_async_le2(); else wait_async_0();
    __syncthreads();

    F16x16 a[4], b[2];
#pragma unroll
    for (int mi = 0; mi < 4; mi++) {
      const _Float16* ap = X + (size_t)(m0 + mi * 16 + ml) * D_ + kb;
      a[mi].h[0] = *(const v8h*)(ap + hi * 8);        // K = kb + hi*8 .. +7
      a[mi].h[1] = *(const v8h*)(ap + 16 + hi * 8);   // K = kb+16+hi*8 .. +7
    }
#pragma unroll
    for (int ni = 0; ni < 2; ni++) {
      const _Float16* bp = &sB[buf][(wn * 32 + ni * 16 + ml) * BSTRIDE + hi * 8];
      b[ni].h[0] = *(const v8h*)(bp);
      b[ni].h[1] = *(const v8h*)(bp + 16);
    }
#pragma unroll
    for (int mi = 0; mi < 4; mi++)
#pragma unroll
      for (int ni = 0; ni < 2; ni++)
        acc[mi][ni] = wmma_f16(a[mi].v, b[ni].v, acc[mi][ni]);
    __syncthreads();   // all waves done reading sB[buf] before it is re-filled
  }

#pragma unroll
  for (int mi = 0; mi < 4; mi++)
#pragma unroll
    for (int ni = 0; ni < 2; ni++)
#pragma unroll
      for (int j = 0; j < 8; j++) {
        int row = m0 + mi * 16 + j + 8 * hi;       // C-layout: M = j + 8*(lane/16)
        int col = nblk + wn * 32 + ni * 16 + ml;   // N = lane%16
        float val = acc[mi][ni][j];
        if (mode == 2) {
          ((float*)Y)[(size_t)row * D_ + col] = val;
        } else {
          int b2 = row >> 11, s = row & (S_ - 1);
          int h  = col >> 6,  d = col & (DK - 1);
          size_t idx = (mode == 0)
              ? ((size_t)(b2 * H_ + h) * S_ + s) * DK + d
              : ((size_t)(b2 * H_ + h) * DK + d) * S_ + s;
          ((_Float16*)Y)[idx] = (_Float16)val;
        }
      }
}

// ---------------- causal flash attention ----------------
// Workgroup = 4 waves = 64 consecutive queries of one (b,h); wave w owns 16 rows.
// K/V tiles (32 keys) fetched ONCE per workgroup into LDS via async loads,
// double-buffered. All waves run the same trip count (barrier-safe); a wave
// skips compute (wave-uniform) for key blocks entirely above its diagonal.
#define KSTR 72   // sK row stride (halves), padded
#define VSTR 40   // sV row stride (halves), padded
__global__ __launch_bounds__(128) void attn_kernel(const _Float16* __restrict__ Qh,
                                                   const _Float16* __restrict__ Kh,
                                                   const _Float16* __restrict__ Vt,
                                                   _Float16* __restrict__ Cc) {
  __shared__ _Float16 sK[2][32 * KSTR];   // keys x dk
  __shared__ _Float16 sV[2][DK * VSTR];   // dk x keys
  __shared__ _Float16 ldsP[4][16 * 32];   // per-wave P tile (C-layout -> A-layout)
  int tid  = threadIdx.x;
  int lane = tid & 31;
  int w    = tid >> 5;
  int bh   = blockIdx.x >> 5;            // (b*16+h)
  int qc   = blockIdx.x & 31;            // 64-query chunk within the sequence
  int qb0  = qc << 6;
  int qw   = qb0 + w * 16;               // this wave's query-tile base
  const _Float16* Q = Qh + (size_t)bh * S_ * DK;
  const _Float16* K = Kh + (size_t)bh * S_ * DK;
  const _Float16* V = Vt + (size_t)bh * DK * S_;  // [dk][S]
  int ml = lane & 15, hi = lane >> 4;

  auto stageKV = [&](int buf, int kb) {
    {  // K tile: 32 rows x 64 halves = 256 x 16B chunks, 1 per thread... x2? 256 chunks
      int row = tid >> 3, ch = tid & 7;  // 8 chunks per row, 16 rows -> 128
      async_ld_b128(K + (size_t)(kb + row) * DK + ch * 8,
                    &sK[buf][row * KSTR + ch * 8]);
    }
    {  // V tile: 64 rows (dk) x 32 halves = 256 x 16B chunks
      int row = tid >> 2, ch = tid & 3;
      async_ld_b128(V + (size_t)row * S_ + kb + ch * 8,
                    &sV[buf][row * VSTR + ch * 8]);
    }
  };

  // Q A-fragments for dk chunks 0..31, 32..63 (kept in regs whole loop)
  F16x16 qf[2];
#pragma unroll
  for (int ks = 0; ks < 2; ks++) {
    const _Float16* qp = Q + (size_t)(qw + ml) * DK + ks * 32;
    qf[ks].h[0] = *(const v8h*)(qp + hi * 8);
    qf[ks].h[1] = *(const v8h*)(qp + 16 + hi * 8);
  }

  float mrow[8], lrow[8];
  v8f O[4];
#pragma unroll
  for (int j = 0; j < 8; j++) { mrow[j] = -1e30f; lrow[j] = 0.f; }
#pragma unroll
  for (int t = 0; t < 4; t++) O[t] = {};

  int nkb = (qb0 >> 5) + 2;  // key blocks covering keys [0, qb0+64)
  stageKV(0, 0);
  int buf = 0;
  for (int i = 0; i < nkb; i++, buf ^= 1) {
    int kb = i * 32;
    bool pre = (i + 1 < nkb);
    if (pre) stageKV(buf ^ 1, kb + 32);
    if (pre) wait_async_le2(); else wait_async_0();
    __syncthreads();

    if (kb <= qw + 15) {  // wave-uniform: skip blocks fully above our diagonal
      // K B-fragments from LDS: B[kd][n] = K[key n][kd]
      F16x16 bk[2][2];  // [key sub-tile nb][dk chunk ks]
#pragma unroll
      for (int nb = 0; nb < 2; nb++)
#pragma unroll
        for (int ks = 0; ks < 2; ks++) {
          const _Float16* kp = &sK[buf][(nb * 16 + ml) * KSTR + ks * 32 + hi * 8];
          bk[nb][ks].h[0] = *(const v8h*)(kp);
          bk[nb][ks].h[1] = *(const v8h*)(kp + 16);
        }
      v8f s0 = {}, s1 = {};
      s0 = wmma_f16(qf[0].v, bk[0][0].v, s0);
      s0 = wmma_f16(qf[1].v, bk[0][1].v, s0);
      s1 = wmma_f16(qf[0].v, bk[1][0].v, s1);
      s1 = wmma_f16(qf[1].v, bk[1][1].v, s1);

      // scale (1/sqrt(64)=0.125), causal mask, online softmax
#pragma unroll
      for (int j = 0; j < 8; j++) {
        int qg = qw + j + 8 * hi;
        float x0 = (kb + ml      > qg) ? -1e9f : s0[j] * 0.125f;
        float x1 = (kb + 16 + ml > qg) ? -1e9f : s1[j] * 0.125f;
        float bm   = red_max16(fmaxf(x0, x1));
        float mnew = fmaxf(mrow[j], bm);
        float alpha = __expf(mrow[j] - mnew);
        float p0 = __expf(x0 - mnew);
        float p1 = __expf(x1 - mnew);
        float rs = red_sum16(p0 + p1);
        lrow[j] = lrow[j] * alpha + rs;
        mrow[j] = mnew;
#pragma unroll
        for (int t = 0; t < 4; t++) O[t][j] = O[t][j] * alpha;
        int r = j + 8 * hi;
        ldsP[w][r * 32 + ml]      = (_Float16)p0;
        ldsP[w][r * 32 + 16 + ml] = (_Float16)p1;
      }
      // per-wave LDS round-trip: wait for DS stores, reload P in A-layout
      asm volatile("s_wait_dscnt 0x0" ::: "memory");
      F16x16 pf;
      pf.h[0] = *(const v8h*)&ldsP[w][ml * 32 + hi * 8];
      pf.h[1] = *(const v8h*)&ldsP[w][ml * 32 + 16 + hi * 8];

      // O += P(16x32) @ V(32x64): B columns = Vt rows in LDS
#pragma unroll
      for (int t = 0; t < 4; t++) {
        const _Float16* vp = &sV[buf][(t * 16 + ml) * VSTR + hi * 8];
        F16x16 vf;
        vf.h[0] = *(const v8h*)(vp);
        vf.h[1] = *(const v8h*)(vp + 16);
        O[t] = wmma_f16(pf.v, vf.v, O[t]);
      }
    }
    __syncthreads();  // all waves done with sK/sV[buf] before re-fill
  }

  int b = bh >> 4, h = bh & 15;
#pragma unroll
  for (int j = 0; j < 8; j++) {
    float inv = 1.f / lrow[j];
    int s = qw + j + 8 * hi;
#pragma unroll
    for (int t = 0; t < 4; t++) {
      int dcol = h * DK + t * 16 + ml;
      Cc[(size_t)(b * S_ + s) * D_ + dcol] = (_Float16)(O[t][j] * inv);
    }
  }
}

extern "C" void kernel_launch(void* const* d_in, const int* in_sizes, int n_in,
                              void* d_out, int out_size, void* d_ws, size_t ws_size,
                              hipStream_t stream) {
  (void)in_sizes; (void)n_in; (void)out_size; (void)ws_size;
  const float* q  = (const float*)d_in[0];
  const float* k  = (const float*)d_in[1];
  const float* v  = (const float*)d_in[2];
  const float* Wq = (const float*)d_in[3];
  const float* Wk = (const float*)d_in[4];
  const float* Wv = (const float*)d_in[5];
  const float* Wo = (const float*)d_in[6];

  const size_t XSZ = (size_t)M_ * D_;          // 8,388,608
  const size_t WSZ = (size_t)D_ * D_;          // 1,048,576
  const size_t HSZ = (size_t)B_ * H_ * S_ * DK;

  _Float16* base = (_Float16*)d_ws;            // ~120 MB total
  _Float16* cq = base;
  _Float16* ck = cq + XSZ;
  _Float16* cv = ck + XSZ;
  _Float16* wq = cv + XSZ;
  _Float16* wk = wq + WSZ;
  _Float16* wv = wk + WSZ;
  _Float16* wo = wv + WSZ;
  _Float16* Qh = wo + WSZ;
  _Float16* Kh = Qh + HSZ;
  _Float16* Vt = Kh + HSZ;
  _Float16* Cc = Vt + HSZ;

  cvt_kernel<<<2048, 256, 0, stream>>>(q,  cq, (int)(XSZ / 4));
  cvt_kernel<<<2048, 256, 0, stream>>>(k,  ck, (int)(XSZ / 4));
  cvt_kernel<<<2048, 256, 0, stream>>>(v,  cv, (int)(XSZ / 4));
  cvt_kernel<<<512, 256, 0, stream>>>(Wq, wq, (int)(WSZ / 4));
  cvt_kernel<<<512, 256, 0, stream>>>(Wk, wk, (int)(WSZ / 4));
  cvt_kernel<<<512, 256, 0, stream>>>(Wv, wv, (int)(WSZ / 4));
  cvt_kernel<<<512, 256, 0, stream>>>(Wo, wo, (int)(WSZ / 4));

  dim3 gg(D_ / 128, M_ / 128);  // (8, 64)
  gemm_xwt<<<gg, 256, 0, stream>>>(cq, wq, Qh, 0);
  gemm_xwt<<<gg, 256, 0, stream>>>(ck, wk, Kh, 0);
  gemm_xwt<<<gg, 256, 0, stream>>>(cv, wv, Vt, 1);

  // one block = 64 queries of one head: B*H*(S/64) = 2048 blocks
  attn_kernel<<<B_ * H_ * (S_ / 64), 128, 0, stream>>>(Qh, Kh, Vt, Cc);

  gemm_xwt<<<gg, 256, 0, stream>>>(Cc, wo, d_out, 2);
}
